// HyperResidualBlock_42167988912490
// MI455X (gfx1250) — compile-verified, matrix operand order checked by reading, not run on hardware
//
#include <hip/hip_runtime.h>

typedef unsigned short u16;
typedef unsigned int   u32;

typedef __attribute__((ext_vector_type(16))) __bf16 v16bf;
typedef __attribute__((ext_vector_type(8)))  float  v8f;

#define B_    16
#define CIN   128
#define COUT  128
#define ZD    128
#define H_    64
#define W_    64
#define HW    (H_ * W_)            // 4096
#define PW    66                   // padded width  (1px halo)
#define PH    66                   // padded height
#define PHW   (PH * PW)            // 4356
#define KKTOT (CIN * 9)            // 1152
#define NHEAD (COUT * CIN * 9)     // 147456
#define NGRP  8
#define GRPN  ((COUT / NGRP) * HW) // 65536
#define NTOT  (B_ * COUT * HW)     // 8388608
#define NPLANE (B_ * CIN)          // 2048

__device__ __forceinline__ u16 f2bf(float f) {
  u32 u = __float_as_uint(f);
  u32 r = u + 0x7FFFu + ((u >> 16) & 1u);   // round-to-nearest-even
  return (u16)(r >> 16);
}

// ------------------------------------------- fp32 -> bf16 with 1px zero halo
__global__ __launch_bounds__(256) void cvt_pad_bf16_kernel(const float* __restrict__ in,
                                                           u16* __restrict__ out) {
  int idx   = blockIdx.x * 256 + threadIdx.x;
  int plane = blockIdx.y;
  if (idx >= PHW) return;
  int yy = idx / PW;
  int xx = idx - yy * PW;
  u16 o = 0;
  if (yy >= 1 && yy <= H_ && xx >= 1 && xx <= W_)
    o = f2bf(in[(size_t)plane * HW + (yy - 1) * W_ + (xx - 1)]);
  out[(size_t)plane * PHW + idx] = o;
}

// ------------------------------------------------- hypernet: w[b][j] = z@hw + hb
// j = o*1152 + (i*9 + kh*3 + kw)  -> layout [b][o][k], k contiguous (conv-friendly)
__global__ __launch_bounds__(256) void hyper_w_kernel(const float* __restrict__ z,
                                                      const float* __restrict__ hw,
                                                      const float* __restrict__ hb,
                                                      u16* __restrict__ wout) {
  __shared__ float sz[B_ * ZD];
  for (int i = threadIdx.x; i < B_ * ZD; i += 256) sz[i] = z[i];
  __syncthreads();

  int j = (blockIdx.x * 256 + threadIdx.x) * 2;    // two columns per thread
  float2 acc[B_];
#pragma unroll
  for (int b = 0; b < B_; ++b) { acc[b].x = 0.f; acc[b].y = 0.f; }

  for (int zd = 0; zd < ZD; ++zd) {
    float2 h = *(const float2*)(hw + (size_t)zd * NHEAD + j);
#pragma unroll
    for (int b = 0; b < B_; ++b) {
      float zz = sz[b * ZD + zd];
      acc[b].x = fmaf(zz, h.x, acc[b].x);          // pairs -> v_pk_fma_f32
      acc[b].y = fmaf(zz, h.y, acc[b].y);
    }
  }
  float2 bias = *(const float2*)(hb + j);
#pragma unroll
  for (int b = 0; b < B_; ++b) {
    u32 pk = (u32)f2bf(acc[b].x + bias.x) | ((u32)f2bf(acc[b].y + bias.y) << 16);
    *(u32*)(wout + (size_t)b * NHEAD + j) = pk;
  }
}

// --------------------------------------------- implicit-GEMM 3x3 conv via WMMA
// out[b][m][y][x] = sum_k W[b][m][k] * Im[k][(y,x)],  k = i*9 + kh*3 + kw
// input is bf16 with 1px halo -> no bounds checks in the gather.
#define WPITCH 40   // halves per LDS row: 32 data + 8 pad (80B pitch, conflict-free)

union Frag { uint4 u[2]; v16bf v; };

__global__ __launch_bounds__(256) void conv3x3_wmma_kernel(const u16* __restrict__ inb,
                                                           const u16* __restrict__ wb,
                                                           float* __restrict__ outp) {
  __shared__ alignas(16) u16 sW[2][COUT * WPITCH];   // [m][k]  2 x 10.0 KB
  __shared__ alignas(16) u16 sI[2][W_   * WPITCH];   // [n][k]  2 x  5.0 KB

  const int tid  = threadIdx.x;
  const int b    = blockIdx.y;
  const int y    = blockIdx.x;                    // one image row = 64-pixel N tile
  const int wave = tid >> 5;                      // 8 waves: 16 out-ch each
  const int lane = tid & 31;

  const u16* wbase = wb  + (size_t)b * (COUT * KKTOT);
  const u16* ibase = inb + (size_t)b * (CIN * PHW);

  v8f acc0 = {}, acc1 = {}, acc2 = {}, acc3 = {};

  // cooperative load roles
  const int wm    = tid >> 1;          // 0..127 : out-channel row of weight chunk
  const int wh    = (tid & 1) * 16;    // half-offset within 32-k chunk
  const int in_n  = tid & 63;          // pixel x within row
  const int in_kr = (tid >> 6) * 8;    // 8-k run for im2col

  const int lrow = lane & 15;
  const int ksel = (lane >> 4) * 8;    // lanes 0-15: k {0-7,16-23}; 16-31: {8-15,24-31}

  // im2col chunk state: decomposition of k0 = next_kc + in_kr into (channel ci, tap r)
  int ci = in_kr / 9;
  int r  = in_kr - 9 * ci;

  uint4 wr0, wr1, ir;

  auto load_chunk = [&](int kc_next) {
    // weights: coalesced 16B loads into regs
    const u16* src = wbase + wm * KKTOT + kc_next + wh;
    wr0 = *(const uint4*)(src);
    wr1 = *(const uint4*)(src + 8);
    __builtin_prefetch(src + 32, 0, 3);           // global_prefetch_b8 (next chunk)
    // im2col: 8 consecutive k from padded image; incremental tap walk, no branches
    int kh = (r >= 6) ? 2 : (r >= 3) ? 1 : 0;
    int kw = r - 3 * kh;
    int rr = r;
    int addr = ci * PHW + (y + kh) * PW + (in_n + kw);
    u32 pk[4];
#pragma unroll
    for (int kp = 0; kp < 4; ++kp) {
      u32 v2 = 0;
#pragma unroll
      for (int h = 0; h < 2; ++h) {
        v2 |= ((u32)ibase[addr]) << (16 * h);
        if (rr == 8)      { rr = 0; kw = 0; addr += PHW - 2 * PW - 2; } // next channel
        else if (kw == 2) { kw = 0; ++rr;  addr += PW - 2; }            // next tap row
        else              { ++kw; ++rr;    ++addr; }                    // next tap col
      }
      pk[kp] = v2;
    }
    ir.x = pk[0]; ir.y = pk[1]; ir.z = pk[2]; ir.w = pk[3];
    ci += 3; r += 5; if (r >= 9) { r -= 9; ++ci; }  // advance state by 32 k
  };

  auto store_chunk = [&](int buf) {
    uint4* dw = (uint4*)(&sW[buf][wm * WPITCH + wh]);
    dw[0] = wr0; dw[1] = wr1;
    *(uint4*)(&sI[buf][in_n * WPITCH + in_kr]) = ir;
  };

  load_chunk(0);
  store_chunk(0);

  int buf = 0;
  for (int kc = 0; kc < KKTOT; kc += 32) {
    __syncthreads();                     // chunk `buf` visible; other buffer free
    const bool more = (kc + 32) < KKTOT;
    if (more) load_chunk(kc + 32);       // global -> regs, overlapped with WMMA

    Frag a;
    {
      const u16* ap = &sW[buf][(wave * 16 + lrow) * WPITCH + ksel];
      a.u[0] = *(const uint4*)(ap);
      a.u[1] = *(const uint4*)(ap + 16);
    }
    Frag b0, b1, b2, b3;
    {
      const u16* bp;
      bp = &sI[buf][(0 * 16 + lrow) * WPITCH + ksel]; b0.u[0] = *(const uint4*)bp; b0.u[1] = *(const uint4*)(bp + 16);
      bp = &sI[buf][(1 * 16 + lrow) * WPITCH + ksel]; b1.u[0] = *(const uint4*)bp; b1.u[1] = *(const uint4*)(bp + 16);
      bp = &sI[buf][(2 * 16 + lrow) * WPITCH + ksel]; b2.u[0] = *(const uint4*)bp; b2.u[1] = *(const uint4*)(bp + 16);
      bp = &sI[buf][(3 * 16 + lrow) * WPITCH + ksel]; b3.u[0] = *(const uint4*)bp; b3.u[1] = *(const uint4*)(bp + 16);
    }
    acc0 = __builtin_amdgcn_wmma_f32_16x16x32_bf16(false, a.v, false, b0.v, (short)0, acc0, false, false);
    acc1 = __builtin_amdgcn_wmma_f32_16x16x32_bf16(false, a.v, false, b1.v, (short)0, acc1, false, false);
    acc2 = __builtin_amdgcn_wmma_f32_16x16x32_bf16(false, a.v, false, b2.v, (short)0, acc2, false, false);
    acc3 = __builtin_amdgcn_wmma_f32_16x16x32_bf16(false, a.v, false, b3.v, (short)0, acc3, false, false);

    if (more) store_chunk(buf ^ 1);      // regs -> other LDS buffer
    buf ^= 1;
  }

  // D layout: vgpr v, lanes 0-15 -> M=v, lanes 16-31 -> M=8+v, N=lane&15
  float* op = outp + (size_t)b * COUT * HW + y * W_;
  const int mb = wave * 16 + ksel;
  const int n0 = lrow;
#pragma unroll
  for (int v = 0; v < 8; ++v) {
    float* row = op + (size_t)(mb + v) * HW + n0;
    row[0]  = acc0[v];
    row[16] = acc1[v];
    row[32] = acc2[v];
    row[48] = acc3[v];
  }
}

// ------------------------------------------------------------ GroupNorm stats
__global__ __launch_bounds__(256) void gn_stats_kernel(const float* __restrict__ in,
                                                       float* __restrict__ stats) {
  const int bg = blockIdx.x;                   // b*8+g ; group = 16 ch contiguous
  const float* p = in + (size_t)bg * GRPN;
  float s = 0.f, ss = 0.f;
  for (int i = threadIdx.x; i < GRPN; i += 256) {
    float v = p[i];
    s += v; ss += v * v;
  }
  __shared__ float rs[256], rq[256];
  rs[threadIdx.x] = s; rq[threadIdx.x] = ss;
  __syncthreads();
  for (int o = 128; o > 0; o >>= 1) {
    if (threadIdx.x < o) { rs[threadIdx.x] += rs[threadIdx.x + o]; rq[threadIdx.x] += rq[threadIdx.x + o]; }
    __syncthreads();
  }
  if (threadIdx.x == 0) {
    float m   = rs[0] * (1.f / GRPN);
    float var = rq[0] * (1.f / GRPN) - m * m;
    stats[bg * 2]     = m;
    stats[bg * 2 + 1] = rsqrtf(var + 1e-5f);
  }
}

// ------------------------ GN + ReLU -> bf16 with 1px zero halo (between convs)
__global__ __launch_bounds__(256) void gn_relu_pad_bf16_kernel(const float* __restrict__ in,
                                                               const float* __restrict__ stats,
                                                               const float* __restrict__ gamma,
                                                               const float* __restrict__ beta,
                                                               u16* __restrict__ outb) {
  int idx   = blockIdx.x * 256 + threadIdx.x;
  int plane = blockIdx.y;                      // b*128 + c
  if (idx >= PHW) return;
  int yy = idx / PW;
  int xx = idx - yy * PW;
  u16 o = 0;
  if (yy >= 1 && yy <= H_ && xx >= 1 && xx <= W_) {
    int c  = plane & (COUT - 1);
    int bg = plane >> 4;                       // (b*128+c)/16 == b*8 + c/16
    float mean = stats[bg * 2], rstd = stats[bg * 2 + 1];
    float v = in[(size_t)plane * HW + (yy - 1) * W_ + (xx - 1)];
    o = f2bf(fmaxf(fmaf((v - mean) * rstd, gamma[c], beta[c]), 0.f));
  }
  outb[(size_t)plane * PHW + idx] = o;
}

// ------------------------------------- GN + residual + ReLU -> fp32 final output
__global__ __launch_bounds__(256) void gn_res_relu_kernel(const float* __restrict__ in,
                                                          const float* __restrict__ stats,
                                                          const float* __restrict__ gamma,
                                                          const float* __restrict__ beta,
                                                          const float* __restrict__ xres,
                                                          float* __restrict__ out) {
  int i4 = blockIdx.x * 256 + threadIdx.x;
  int e0 = i4 * 4;
  int c  = (e0 >> 12) & 127;
  int bg = e0 >> 16;
  float mean = stats[bg * 2], rstd = stats[bg * 2 + 1];
  float g = gamma[c], bt = beta[c];
  float4 v  = ((const float4*)in)[i4];
  float4 xr = ((const float4*)xres)[i4];
  float4 o;
  o.x = fmaxf(fmaf((v.x - mean) * rstd, g, bt) + xr.x, 0.f);
  o.y = fmaxf(fmaf((v.y - mean) * rstd, g, bt) + xr.y, 0.f);
  o.z = fmaxf(fmaf((v.z - mean) * rstd, g, bt) + xr.z, 0.f);
  o.w = fmaxf(fmaf((v.w - mean) * rstd, g, bt) + xr.w, 0.f);
  ((float4*)out)[i4] = o;
}

extern "C" void kernel_launch(void* const* d_in, const int* in_sizes, int n_in,
                              void* d_out, int out_size, void* d_ws, size_t ws_size,
                              hipStream_t stream) {
  const float* x    = (const float*)d_in[0];
  const float* z    = (const float*)d_in[1];
  const float* h1w  = (const float*)d_in[2];
  const float* h1b  = (const float*)d_in[3];
  const float* h2w  = (const float*)d_in[4];
  const float* h2b  = (const float*)d_in[5];
  const float* g1   = (const float*)d_in[6];
  const float* bt1  = (const float*)d_in[7];
  const float* g2   = (const float*)d_in[8];
  const float* bt2  = (const float*)d_in[9];
  float* out = (float*)d_out;

  char* ws = (char*)d_ws;
  size_t off = 0;
  u16*   xpad    = (u16*)(ws + off); off += (size_t)NPLANE * PHW * 2;         // 17.8 MB
  u16*   a1pad   = (u16*)(ws + off); off += (size_t)NPLANE * PHW * 2;         // 17.8 MB
  u16*   w1      = (u16*)(ws + off); off += (size_t)B_ * NHEAD * 2;           // 4.7 MB
  u16*   w2      = (u16*)(ws + off); off += (size_t)B_ * NHEAD * 2;           // 4.7 MB
  float* convbuf = (float*)(ws + off); off += (size_t)NTOT * 4;               // 33.5 MB
  float* stats   = (float*)(ws + off); off += (size_t)B_ * NGRP * 2 * 4;

  const dim3 PAD_GRID((PHW + 255) / 256, NPLANE);  // 18 x 2048
  const int  ELT4_BLOCKS = NTOT / 4 / 256;         // 8192
  const int  HYP_BLOCKS  = NHEAD / (256 * 2);      // 288

  // 1) x -> padded bf16 (zero halo)
  cvt_pad_bf16_kernel<<<PAD_GRID, 256, 0, stream>>>(x, xpad);
  // 2) hypernet weight generation (fp32 GEMM, bf16 output)
  hyper_w_kernel<<<HYP_BLOCKS, 256, 0, stream>>>(z, h1w, h1b, w1);
  hyper_w_kernel<<<HYP_BLOCKS, 256, 0, stream>>>(z, h2w, h2b, w2);
  // 3) conv1 (WMMA implicit GEMM, double-buffered LDS)
  conv3x3_wmma_kernel<<<dim3(H_, B_), 256, 0, stream>>>(xpad, w1, convbuf);
  // 4) GN1 stats + apply(+ReLU) -> padded bf16
  gn_stats_kernel<<<B_ * NGRP, 256, 0, stream>>>(convbuf, stats);
  gn_relu_pad_bf16_kernel<<<PAD_GRID, 256, 0, stream>>>(convbuf, stats, g1, bt1, a1pad);
  // 5) conv2
  conv3x3_wmma_kernel<<<dim3(H_, B_), 256, 0, stream>>>(a1pad, w2, convbuf);
  // 6) GN2 stats + apply + residual + ReLU -> fp32 out
  gn_stats_kernel<<<B_ * NGRP, 256, 0, stream>>>(convbuf, stats);
  gn_res_relu_kernel<<<ELT4_BLOCKS, 256, 0, stream>>>(convbuf, stats, g2, bt2, x, out);
}
